// GLoCELayerOutProp_10917806867028
// MI455X (gfx1250) — matrix operand order repaired and checked by reading, not compiled, not run on hardware
//
#include <hip/hip_runtime.h>

// ---------------- problem constants ----------------
#define DMODEL 2048
#define NCON   8
#define SRANK  8
#define HRANK  8
#define NTOK   4096            // B*T = 4*1024
#define COLS   160             // 64 P(n,s) + 8 Q(n) + 64 R(n,h) + 24 pad (10 tiles)
#define KCHUNK 128             // K staged in LDS per iteration
#define NTILE  5               // col tiles per wave (compile-time!)

typedef __attribute__((ext_vector_type(16))) __bf16 v16bf;
typedef __attribute__((ext_vector_type(8)))  __bf16 v8bf;
typedef __attribute__((ext_vector_type(8)))  float  v8f;

// ---------------------------------------------------------------------------
// K1: tiny constant reductions.
//   cmat[n*8+s]  = sum_d select_weight[n,d,s] * mean_diff[n,d]
//   mnorm[n]     = sum_d mean_diff[n,d]^2
//   dc[n*8+h]    = sum_d debias_w[n,d] * lora_update[n,d,h]
// one wave per output, lane-strided dot + shuffle reduce (wave32).
// ---------------------------------------------------------------------------
__global__ void gloce_precompute(const float* __restrict__ sw,
                                 const float* __restrict__ md,
                                 const float* __restrict__ dbw,
                                 const float* __restrict__ upd,
                                 float* __restrict__ cmat,
                                 float* __restrict__ mnorm,
                                 float* __restrict__ dc) {
  const int wave = threadIdx.x >> 5;
  const int lane = threadIdx.x & 31;
  for (int o = wave; o < 136; o += 8) {
    float sum = 0.0f;
    if (o < 64) {
      const int n = o >> 3, s = o & 7;
      const size_t base = (size_t)n * DMODEL;
      for (int d = lane; d < DMODEL; d += 32)
        sum += sw[(base + d) * SRANK + s] * md[base + d];
    } else if (o < 72) {
      const int n = o - 64;
      const size_t base = (size_t)n * DMODEL;
      for (int d = lane; d < DMODEL; d += 32) {
        float v = md[base + d];
        sum += v * v;
      }
    } else {
      const int i = o - 72, n = i >> 3, h = i & 7;
      const size_t base = (size_t)n * DMODEL;
      for (int d = lane; d < DMODEL; d += 32)
        sum += dbw[base + d] * upd[(base + d) * HRANK + h];
    }
    for (int off = 16; off > 0; off >>= 1) sum += __shfl_down(sum, off, 32);
    if (lane == 0) {
      if (o < 64)      cmat[o] = sum;
      else if (o < 72) mnorm[o - 64] = sum;
      else             dc[o - 72] = sum;
    }
  }
}

// ---------------------------------------------------------------------------
// K1b: pack the 136 weight columns into bf16, column-major in K:
//   Wb[col][k], col 0..63 = select_weight(n,s), 64..71 = mean_diff(n),
//   72..135 = lora_update(n,h), 136..159 = zero padding.
// ---------------------------------------------------------------------------
__global__ void gloce_pack(const float* __restrict__ sw,
                           const float* __restrict__ md,
                           const float* __restrict__ upd,
                           __bf16* __restrict__ Wb) {
  const int idx = blockIdx.x * 256 + threadIdx.x;
  if (idx >= COLS * DMODEL) return;
  const int col = idx >> 11;        // / DMODEL
  const int d   = idx & (DMODEL - 1);
  float v = 0.0f;
  if (col < 64) {
    const int n = col >> 3, s = col & 7;
    v = sw[((size_t)n * DMODEL + d) * SRANK + s];
  } else if (col < 72) {
    const int n = col - 64;
    v = md[(size_t)n * DMODEL + d];
  } else if (col < 136) {
    const int i = col - 72, n = i >> 3, h = i & 7;
    v = upd[((size_t)n * DMODEL + d) * HRANK + h];
  }
  Wb[idx] = (__bf16)v;
}

// ---------------------------------------------------------------------------
// K2: fused selector+LoRA-projection GEMM (bf16 WMMA, f32 accumulate).
//   P[token][0..159] = x[token] . Wb[:,col]
// Block = 256 threads = 8 waves handling 4 row tiles (64 tokens) x 10 col
// tiles. Wave w: row tile (w&3), column group (w>>2), each group owns a
// compile-time-constant 5 tiles -> the jj loop fully unrolls and the 40
// accumulator VGPRs stay in registers (no runtime-indexed acc array,
// no scratch). B chunks (160 x 128 bf16 = 40 KB) staged in LDS; the two
// waves of a row tile re-read the same x lines (WGP$-hit).
// ---------------------------------------------------------------------------
__global__ __launch_bounds__(256)
void gloce_selector_gemm(const float* __restrict__ x,
                         const __bf16* __restrict__ Wb,
                         float* __restrict__ P) {
  __shared__ __bf16 ldsB[COLS * KCHUNK];   // [col][k_local]

  const int lane  = threadIdx.x & 31;
  const int wave  = threadIdx.x >> 5;
  const int rt    = wave & 3;                                   // row tile
  const int cg    = __builtin_amdgcn_readfirstlane(wave >> 2);  // col group (sgpr)
  const int jbase = cg * NTILE;            // runtime base, compile-time count
  const int m     = lane & 15;             // row within tile / col within tile
  const int khalf = lane >> 4;             // K-half select (contiguous k order)
  const int rowbase = (blockIdx.x * 4 + rt) * 16;
  const float* __restrict__ xrow = x + (size_t)(rowbase + m) * DMODEL;
  // this wave's first B-fragment address inside the LDS chunk
  const __bf16* __restrict__ ldsBase =
      ldsB + (jbase * 16 + m) * KCHUNK + khalf * 16;

  v8f acc[NTILE];
  const v8f vzero = {0.f, 0.f, 0.f, 0.f, 0.f, 0.f, 0.f, 0.f};
#pragma unroll
  for (int jj = 0; jj < NTILE; ++jj) acc[jj] = vzero;

  for (int k0 = 0; k0 < DMODEL; k0 += KCHUNK) {
    __syncthreads();
    // cooperative B-chunk stage: 2560 granules of 16B, 10 per thread
    for (int g = threadIdx.x; g < (COLS * KCHUNK) / 8; g += 256) {
      const int col = g >> 4;              // 16 granules of 8 elems per col
      const int off = (g & 15) * 8;
      *(v8bf*)(ldsB + col * KCHUNK + off) =
          *(const v8bf*)(Wb + (size_t)col * DMODEL + k0 + off);
    }
    __syncthreads();

    if (k0 + KCHUNK < DMODEL)              // pull next x chunk toward the WGP
      __builtin_prefetch(xrow + k0 + KCHUNK, 0, 1);

#pragma unroll
    for (int kk = 0; kk < KCHUNK; kk += 32) {
      // ---- A fragment (16-bit A 16x32, contiguous k = 16*half + i order):
      // lane half 0 holds K = kk+[0..15]; half 1 holds kk+[16..31] (matches B).
      const float* s0 = xrow + k0 + kk + khalf * 16;
      const float4 f0 = *(const float4*)(s0);
      const float4 f1 = *(const float4*)(s0 + 4);
      const float4 f2 = *(const float4*)(s0 + 8);
      const float4 f3 = *(const float4*)(s0 + 12);
      const float av[16] = {f0.x, f0.y, f0.z, f0.w, f1.x, f1.y, f1.z, f1.w,
                            f2.x, f2.y, f2.z, f2.w, f3.x, f3.y, f3.z, f3.w};
      v16bf a;
#pragma unroll
      for (int i = 0; i < 16; ++i) a[i] = (__bf16)av[i];
      // ---- 5 col tiles: B fragment = 16 contiguous bf16 (32B LDS read)
#pragma unroll
      for (int jj = 0; jj < NTILE; ++jj) {
        const v16bf b = *(const v16bf*)(ldsBase + jj * 16 * KCHUNK + kk);
        acc[jj] = __builtin_amdgcn_wmma_f32_16x16x32_bf16(
            false, a, false, b, (short)0, acc[jj], false, false);
      }
    }
  }

  // ---- store C tiles: lane holds col (jt*16+m), rows 8*khalf + v
#pragma unroll
  for (int jj = 0; jj < NTILE; ++jj) {
    const int jt = jbase + jj;
#pragma unroll
    for (int v = 0; v < 8; ++v) {
      const int row = rowbase + khalf * 8 + v;
      P[(size_t)row * COLS + jt * 16 + m] = acc[jj][v];
    }
  }
}

// ---------------------------------------------------------------------------
// K3: per-token epilogue (wave per token).
//   xx       = ||x||^2 (lane-strided float4 pass + wave32 xor-reduce)
//   score[n] = sum_s (P[n,s]-cmat[n,s])^2 / (xx - 2 Q[n] + mnorm[n])
//   gate[n]  = sigmoid(slope[n]*(score[n]-center[n])); s = max, c = argmax
//   mod_x[h] = R[c,h] - dc[c,h]
//   out[d]   = (1-s)*x[d] + s*(bias_w[c,d] + dgn[c,d,:].mod_x)
// ---------------------------------------------------------------------------
__global__ void gloce_epilogue(const float* __restrict__ x,
                               const float* __restrict__ P,
                               const float* __restrict__ cmat,
                               const float* __restrict__ mnorm,
                               const float* __restrict__ dc,
                               const float* __restrict__ center,
                               const float* __restrict__ slope,
                               const float* __restrict__ dgn,
                               const float* __restrict__ bw,
                               float* __restrict__ out) {
  const int wave = threadIdx.x >> 5;
  const int lane = threadIdx.x & 31;
  const int token = blockIdx.x * 8 + wave;
  if (token >= NTOK) return;

  const float* __restrict__ xr = x + (size_t)token * DMODEL;

  // ---- ||x||^2 in f32
  float xxp = 0.0f;
#pragma unroll 4
  for (int it = 0; it < DMODEL / 128; ++it) {
    const float4 xv = *(const float4*)(xr + it * 128 + lane * 4);
    xxp += xv.x * xv.x + xv.y * xv.y + xv.z * xv.z + xv.w * xv.w;
  }
#pragma unroll
  for (int off = 16; off > 0; off >>= 1) xxp += __shfl_xor(xxp, off, 32);
  const float xx = xxp;                      // all lanes hold the sum

  // ---- gates / argmax (redundant per lane; tiny, L0-broadcast reads)
  const float* p = P + (size_t)token * COLS;
  float best = -1.0f;
  int   bi   = 0;
#pragma unroll
  for (int n = 0; n < NCON; ++n) {
    const float q   = p[64 + n];
    const float inv = 1.0f / (xx - 2.0f * q + mnorm[n]);
    float sc = 0.0f;
#pragma unroll
    for (int s = 0; s < SRANK; ++s) {
      const float t = p[n * 8 + s] - cmat[n * 8 + s];
      sc += t * t;
    }
    sc *= inv;
    const float z = slope[n] * (sc - center[n]);
    const float g = 1.0f / (1.0f + __expf(-z));
    if (g > best) { best = g; bi = n; }
  }
  const float sgate = best;
  const float one_m = 1.0f - sgate;

  float modx[HRANK];
#pragma unroll
  for (int h = 0; h < HRANK; ++h)
    modx[h] = p[72 + bi * 8 + h] - dc[bi * 8 + h];

  const float* __restrict__ dgc = dgn + (size_t)bi * DMODEL * HRANK;
  const float* __restrict__ bc  = bw  + (size_t)bi * DMODEL;
  float* __restrict__ orow      = out + (size_t)token * DMODEL;

  // ---- blend: each lane owns 4 consecutive d -> dgn reads are one
  // contiguous 128B run per lane per iteration.
#pragma unroll 2
  for (int it = 0; it < DMODEL / 128; ++it) {
    const int d0 = it * 128 + lane * 4;
    const float4* gp = (const float4*)(dgc + (size_t)d0 * 8);
    float od[4];
#pragma unroll
    for (int e = 0; e < 4; ++e) {
      const float4 a = gp[2 * e];
      const float4 b = gp[2 * e + 1];
      od[e] = a.x * modx[0] + a.y * modx[1] + a.z * modx[2] + a.w * modx[3] +
              b.x * modx[4] + b.y * modx[5] + b.z * modx[6] + b.w * modx[7];
    }
    const float4 xv = *(const float4*)(xr + d0);
    const float4 bv = *(const float4*)(bc + d0);
    float4 o;
    o.x = one_m * xv.x + sgate * (bv.x + od[0]);
    o.y = one_m * xv.y + sgate * (bv.y + od[1]);
    o.z = one_m * xv.z + sgate * (bv.z + od[2]);
    o.w = one_m * xv.w + sgate * (bv.w + od[3]);
    *(float4*)(orow + d0) = o;
  }
}

// ---------------------------------------------------------------------------
extern "C" void kernel_launch(void* const* d_in, const int* in_sizes, int n_in,
                              void* d_out, int out_size, void* d_ws, size_t ws_size,
                              hipStream_t stream) {
  (void)in_sizes; (void)n_in; (void)out_size; (void)ws_size;
  const float* x   = (const float*)d_in[0];   // [B,T,D]
  const float* sw  = (const float*)d_in[1];   // [N,D,S]
  const float* md  = (const float*)d_in[2];   // [N,D]
  const float* ic  = (const float*)d_in[3];   // [N]
  const float* is  = (const float*)d_in[4];   // [N]
  const float* upd = (const float*)d_in[5];   // [N,D,H]
  const float* dgn = (const float*)d_in[6];   // [N,D,H]
  const float* bw  = (const float*)d_in[7];   // [N,D]
  const float* dbw = (const float*)d_in[8];   // [N,D]
  float* out = (float*)d_out;

  // workspace layout (~3.3 MB total)
  char* w = (char*)d_ws;
  float*  cmat  = (float*)(w + 0);                        //  64 f32
  float*  mnorm = (float*)(w + 256);                      //   8 f32
  float*  dc    = (float*)(w + 512);                      //  64 f32
  __bf16* Wb    = (__bf16*)(w + 1024);                    // 160*2048 bf16 = 640 KB
  float*  P     = (float*)(w + 1024 + COLS * DMODEL * 2); // 4096*160 f32 = 2.5 MB

  gloce_precompute<<<1, 256, 0, stream>>>(sw, md, dbw, upd, cmat, mnorm, dc);
  gloce_pack<<<(COLS * DMODEL + 255) / 256, 256, 0, stream>>>(sw, md, upd, Wb);
  gloce_selector_gemm<<<NTOK / 64, 256, 0, stream>>>(x, Wb, P);
  gloce_epilogue<<<NTOK / 8, 256, 0, stream>>>(x, P, cmat, mnorm, dc,
                                               ic, is, dgn, bw, out);
}
